// SPDCNFusion_14998025797707
// MI455X (gfx1250) — compile-verified, matrix operand order checked by reading, not run on hardware
//
#include <hip/hip_runtime.h>
#include <math.h>

typedef __attribute__((ext_vector_type(8)))  float  v8f;
typedef __attribute__((ext_vector_type(16))) __bf16 v16bf;

#define Hh   128
#define Ww   128
#define HW   (128 * 128)
#define Kt   9
#define QSTR 68      // LDS row stride (dwords): lane base banks = 4*l15 -> each bank hit exactly 2x per b128 op
#define PT   64      // positions per workgroup (64 | W => one (b,h) row segment)

// ---- f32 <-> bf16 split helpers (RNE, pure bit math; bf16 keeps f32 exponent range) ----
__device__ __forceinline__ unsigned short f2bf(float v) {
    union { float f; unsigned u; } x; x.f = v;
    unsigned r = x.u + 0x7FFFu + ((x.u >> 16) & 1u);   // round-to-nearest-even
    return (unsigned short)(r >> 16);
}
__device__ __forceinline__ float bf2f(unsigned short b) {
    union { float f; unsigned u; } x; x.u = ((unsigned)b) << 16;
    return x.f;
}

union BFrag { uint4 q[2]; v16bf v; };   // two b128 quads == one 16x32/32x16 bf16 fragment

// -------- Kernel 1: 1x1 conv head -> dy, dx, sigmoid(mask) per (b,k,h,w) --------
__global__ __launch_bounds__(256) void offmask_kernel(
    const float* __restrict__ x0, const float* __restrict__ x1,
    const float* __restrict__ w_off, const float* __restrict__ b_off,
    const float* __restrict__ w_mask, const float* __restrict__ b_mask,
    float* __restrict__ dyo, float* __restrict__ dxo, float* __restrict__ mo)
{
    __shared__ float lw[27 * 128];          // 18 offset rows + 9 mask rows, 13.8 KB
    const int tid = threadIdx.x;
    for (int i = tid; i < 27 * 128; i += 256) {
        int j = i >> 7, c = i & 127;
        lw[i] = (j < 18) ? w_off[j * 128 + c] : w_mask[(j - 18) * 128 + c];
    }
    __syncthreads();

    const int p  = blockIdx.x * 256 + tid;  // over B*H*W = 32768
    const int b  = p >> 14;
    const int hw = p & (HW - 1);

    float acc[27];
#pragma unroll
    for (int j = 0; j < 27; ++j) acc[j] = (j < 18) ? b_off[j] : b_mask[j - 18];

    const float* px0 = x0 + (size_t)b * 64 * HW + hw;
    const float* px1 = x1 + (size_t)b * 64 * HW + hw;
    for (int c = 0; c < 128; ++c) {
        float xv = (c < 64) ? px0[c * HW] : px1[(c - 64) * HW];
#pragma unroll
        for (int j = 0; j < 27; ++j) acc[j] = fmaf(xv, lw[j * 128 + c], acc[j]);
    }

#pragma unroll
    for (int k = 0; k < 9; ++k) {
        int o = (b * 9 + k) * HW + hw;
        dyo[o] = acc[2 * k];
        dxo[o] = acc[2 * k + 1];
        mo[o]  = 1.0f / (1.0f + expf(-acc[18 + k]));
    }
}

// -------- Kernel 2: split w_conv into fragment-native bf16 hi/lo layout --------
// Wt[k][s][half][o][j] : per-lane A-fragment = 8 contiguous dwords (2 x b128)
__global__ __launch_bounds__(256) void wsplit_kernel(
    const float* __restrict__ w_conv,
    unsigned* __restrict__ Wt_hi, unsigned* __restrict__ Wt_lo)
{
    int i = blockIdx.x * 256 + threadIdx.x;   // 9*4*2*128*8 = 73728 dwords
    int j    = i & 7;
    int o    = (i >> 3) & 127;
    int half = (i >> 10) & 1;
    int s    = (i >> 11) & 3;
    int k    = i >> 13;
    // ISA 16-bit A/B fragment pair index for dword j of the fragment
    int q    = s * 16 + (j & 3) + 4 * half + 8 * (j >> 2);
    float v0 = w_conv[(o * 128 + 2 * q) * 9 + k];
    float v1 = w_conv[(o * 128 + 2 * q + 1) * 9 + k];
    unsigned short h0 = f2bf(v0), h1 = f2bf(v1);
    unsigned short l0 = f2bf(v0 - bf2f(h0)), l1 = f2bf(v1 - bf2f(h1));
    Wt_hi[i] = (unsigned)h0 | ((unsigned)h1 << 16);   // even K in bits [15:0]
    Wt_lo[i] = (unsigned)l0 | ((unsigned)l1 << 16);
}

// -------- Kernel 3: bilinear gather -> bf16-split LDS tile -> V_WMMA_F32_16X16X32_BF16 (x3) --------
__global__ __launch_bounds__(256) void dcn_wmma_kernel(
    const float* __restrict__ x0, const float* __restrict__ x1,
    const unsigned* __restrict__ Wt_hi, const unsigned* __restrict__ Wt_lo,
    const float* __restrict__ dyo, const float* __restrict__ dxo,
    const float* __restrict__ mo,
    float* __restrict__ out)
{
    // position-major tile: S[pos][K-pair]; per-lane B fragment = 2 aligned b128 runs
    __shared__ alignas(16) unsigned S_hi[PT * QSTR];   // 17.4 KB
    __shared__ alignas(16) unsigned S_lo[PT * QSTR];

    const int tid  = threadIdx.x;
    const int lane = tid & 31;
    const int wv   = tid >> 5;               // 8 waves -> 8 M-tiles of 16
    const int p0   = blockIdx.x * PT;
    const int b    = p0 >> 14;
    const int hw0  = p0 & (HW - 1);
    const int h    = hw0 >> 7;
    const int w0   = hw0 & 127;

    // fill-phase mapping: 64 positions x 4 channel-groups of 32 channels (16 pairs)
    const int fp    = tid & 63;
    const int fg    = tid >> 6;
    const float* src = (fg < 2 ? x0 : x1) + ((size_t)b * 64 + (fg & 1) * 32) * HW;
    const int qbase = fg * 16;
    const int wcol  = w0 + fp;

    // WMMA lane mapping
    const int half = lane >> 4;
    const int l15  = lane & 15;
    const int o0   = wv * 16;

    v8f acc[4];
#pragma unroll
    for (int n = 0; n < 4; ++n) acc[n] = (v8f){0.f, 0.f, 0.f, 0.f, 0.f, 0.f, 0.f, 0.f};

    for (int k = 0; k < 9; ++k) {
        __syncthreads();                     // protect S from previous tap's readers

        // ---- gather/fill S_hi/S_lo for this tap (b128 LDS stores) ----
        {
            const int oidx = (b * 9 + k) * HW + h * Ww + wcol;
            const float dy = dyo[oidx], dx = dxo[oidx], m = mo[oidx];
            const float py = (float)h    - 1.0f + (float)(k / 3) + dy;
            const float px = (float)wcol - 1.0f + (float)(k % 3) + dx;
            const float y0f = floorf(py), x0f = floorf(px);
            const float wy = py - y0f, wx = px - x0f;
            const int y0 = (int)y0f, x0i = (int)x0f;
            const int y1 = y0 + 1,  x1i = x0i + 1;
            const float vy0 = (y0  >= 0 && y0  < Hh) ? 1.f : 0.f;
            const float vy1 = (y1  >= 0 && y1  < Hh) ? 1.f : 0.f;
            const float vx0 = (x0i >= 0 && x0i < Ww) ? 1.f : 0.f;
            const float vx1 = (x1i >= 0 && x1i < Ww) ? 1.f : 0.f;
            const float w00 = m * (1.f - wy) * (1.f - wx) * vy0 * vx0;
            const float w01 = m * (1.f - wy) * wx          * vy0 * vx1;
            const float w10 = m * wy          * (1.f - wx) * vy1 * vx0;
            const float w11 = m * wy          * wx          * vy1 * vx1;
            const int y0c = min(max(y0, 0), Hh - 1), y1c = min(max(y1, 0), Hh - 1);
            const int x0c = min(max(x0i, 0), Ww - 1), x1c = min(max(x1i, 0), Ww - 1);
            const int i00 = y0c * Ww + x0c, i01 = y0c * Ww + x1c;
            const int i10 = y1c * Ww + x0c, i11 = y1c * Ww + x1c;

#pragma unroll
            for (int t = 0; t < 4; ++t) {    // 4 quads of channel pairs
                unsigned bh[4], bl[4];
#pragma unroll
                for (int u = 0; u < 4; ++u) {
                    const int qi = t * 4 + u;
                    const float* sc0 = src + (size_t)(2 * qi)     * HW;
                    const float* sc1 = src + (size_t)(2 * qi + 1) * HW;
                    float v0 = w00 * sc0[i00] + w01 * sc0[i01] + w10 * sc0[i10] + w11 * sc0[i11];
                    float v1 = w00 * sc1[i00] + w01 * sc1[i01] + w10 * sc1[i10] + w11 * sc1[i11];
                    unsigned short h0 = f2bf(v0), h1 = f2bf(v1);
                    unsigned short l0 = f2bf(v0 - bf2f(h0)), l1 = f2bf(v1 - bf2f(h1));
                    bh[u] = (unsigned)h0 | ((unsigned)h1 << 16);
                    bl[u] = (unsigned)l0 | ((unsigned)l1 << 16);
                }
                const int sidx = fp * QSTR + qbase + 4 * t;     // 16B-aligned
                *(uint4*)(&S_hi[sidx]) = make_uint4(bh[0], bh[1], bh[2], bh[3]);
                *(uint4*)(&S_lo[sidx]) = make_uint4(bl[0], bl[1], bl[2], bl[3]);
            }
        }
        __syncthreads();

        // ---- WMMA phase: 4 K-blocks of 32 channels; bf16x3 per block per N-tile ----
#pragma unroll 1
        for (int s = 0; s < 4; ++s) {
            const int qb = s * 16;
            const size_t wi = (size_t)(k * 4 + s) * 2048 + (size_t)half * 1024
                            + (size_t)(o0 + l15) * 8;
            const uint4* pa_h = (const uint4*)(Wt_hi + wi);     // 2 x global b128, coalesced
            const uint4* pa_l = (const uint4*)(Wt_lo + wi);
            BFrag Ahi, Alo;
            Ahi.q[0] = pa_h[0]; Ahi.q[1] = pa_h[1];
            Alo.q[0] = pa_l[0]; Alo.q[1] = pa_l[1];
#pragma unroll
            for (int n = 0; n < 4; ++n) {
                const int col = n * 16 + l15;                   // position within tile
                const unsigned* sh = S_hi + col * QSTR + qb + 4 * half;
                const unsigned* sl = S_lo + col * QSTR + qb + 4 * half;
                BFrag Bhi, Blo;
                Bhi.q[0] = *(const uint4*)(sh);                 // 2 x ds_load_b128 each
                Bhi.q[1] = *(const uint4*)(sh + 8);
                Blo.q[0] = *(const uint4*)(sl);
                Blo.q[1] = *(const uint4*)(sl + 8);
                acc[n] = __builtin_amdgcn_wmma_f32_16x16x32_bf16(
                    false, Ahi.v, false, Bhi.v, (short)0, acc[n], false, false);
                acc[n] = __builtin_amdgcn_wmma_f32_16x16x32_bf16(
                    false, Ahi.v, false, Blo.v, (short)0, acc[n], false, false);
                acc[n] = __builtin_amdgcn_wmma_f32_16x16x32_bf16(
                    false, Alo.v, false, Bhi.v, (short)0, acc[n], false, false);
            }
        }
    }

    // ---- epilogue: ReLU + store (C/D f32 layout: VGPR r -> M=r / M=r+8 by half-wave) ----
#pragma unroll
    for (int n = 0; n < 4; ++n) {
#pragma unroll
        for (int r = 0; r < 8; ++r) {
            const int o = o0 + r + half * 8;
            float v = acc[n][r];
            v = v > 0.f ? v : 0.f;
            out[((size_t)b * 128 + o) * HW + h * Ww + w0 + n * 16 + l15] = v;
        }
    }
}

extern "C" void kernel_launch(void* const* d_in, const int* in_sizes, int n_in,
                              void* d_out, int out_size, void* d_ws, size_t ws_size,
                              hipStream_t stream) {
    const float* x0     = (const float*)d_in[0];
    const float* x1     = (const float*)d_in[1];
    const float* w_off  = (const float*)d_in[2];
    const float* b_off  = (const float*)d_in[3];
    const float* w_mask = (const float*)d_in[4];
    const float* b_mask = (const float*)d_in[5];
    const float* w_conv = (const float*)d_in[6];
    float* out = (float*)d_out;

    // workspace layout
    float* ws  = (float*)d_ws;
    const size_t nOff = (size_t)2 * Kt * HW;       // 294912 floats per array
    float*    ws_dy = ws;
    float*    ws_dx = ws + nOff;
    float*    ws_m  = ws + 2 * nOff;
    unsigned* ws_wh = (unsigned*)(ws + 3 * nOff);  // 73728 dwords (16B-aligned offset)
    unsigned* ws_wl = ws_wh + (size_t)Kt * 4 * 2 * 128 * 8;

    // 1) offsets + masks
    offmask_kernel<<<128, 256, 0, stream>>>(x0, x1, w_off, b_off, w_mask, b_mask,
                                            ws_dy, ws_dx, ws_m);
    // 2) weight split into fragment-native layout: 73728 / 256
    wsplit_kernel<<<288, 256, 0, stream>>>(w_conv, ws_wh, ws_wl);
    // 3) fused gather + bf16x3 WMMA GEMM
    dcn_wmma_kernel<<<512, 256, 0, stream>>>(x0, x1, ws_wh, ws_wl,
                                             ws_dy, ws_dx, ws_m, out);
}
